// VectorQuantizer_79551384256864
// MI455X (gfx1250) — matrix operand order repaired and checked by reading, not compiled, not run on hardware
//
#include <hip/hip_runtime.h>

typedef __bf16 bf16;
typedef __attribute__((ext_vector_type(8)))  __bf16 v8bf;
typedef __attribute__((ext_vector_type(16))) __bf16 v16bf;
typedef __attribute__((ext_vector_type(8)))  float  v8f;

#define D_DIM 256
#define K_VEC 1024
#define N_ROWS 32768
#define MB 32          /* rows per block */
#define HWSZ 1024      /* H*W */

__device__ __forceinline__ v16bf cat8(v8bf a, v8bf b) {
  return __builtin_shufflevector(a, b, 0,1,2,3,4,5,6,7,8,9,10,11,12,13,14,15);
}

// ---------------- codebook prep: bf16 hi/lo split + ||e||^2 ----------------
__global__ __launch_bounds__(256) void vq_prep(const float* __restrict__ cb,
                                               bf16* __restrict__ cb_hi,
                                               bf16* __restrict__ cb_lo,
                                               float* __restrict__ esq) {
  __shared__ float sred[8];
  const int k = blockIdx.x, t = threadIdx.x;
  float v = cb[k * D_DIM + t];
  bf16 h = (bf16)v;
  bf16 l = (bf16)(v - (float)h);
  cb_hi[k * D_DIM + t] = h;
  cb_lo[k * D_DIM + t] = l;
  float s = v * v;
  #pragma unroll
  for (int m = 16; m >= 1; m >>= 1) s += __shfl_xor(s, m);
  if ((t & 31) == 0) sred[t >> 5] = s;
  __syncthreads();
  if (t == 0) {
    float tot = 0.f;
    #pragma unroll
    for (int i = 0; i < 8; ++i) tot += sred[i];
    esq[k] = tot;
  }
}

// ---------------- main fused kernel ----------------
__global__ __launch_bounds__(256) void vq_main(
    const float* __restrict__ x, const float* __restrict__ cb,
    const bf16* __restrict__ cb_hi, const bf16* __restrict__ cb_lo,
    const float* __restrict__ esq,
    float* __restrict__ out, float* __restrict__ loss_part) {

  __shared__ alignas(16) bf16 sA_hi[MB][D_DIM + 8];
  __shared__ alignas(16) bf16 sA_lo[MB][D_DIM + 8];
  __shared__ float s_cval[8][MB];
  __shared__ int   s_cidx[8][MB];
  __shared__ int   s_fidx[MB];
  __shared__ float s_red[8];

  const int tid  = threadIdx.x;
  const int wv   = tid >> 5;
  const int lane = tid & 31;
  const int g    = lane >> 4;   // lane half
  const int rc   = lane & 15;   // row (A) / col (B) within tile

  const int n0  = blockIdx.x * MB;   // first flat row (b*1024 + hw)
  const int b   = n0 >> 10;
  const int hw0 = n0 & 1023;

  // ---- stage x tile (32 rows x 256 dims) into LDS as bf16 hi/lo ----
  {
    const float* xp = x + (size_t)b * (D_DIM * HWSZ) + (size_t)tid * HWSZ + hw0;
    #pragma unroll
    for (int r4 = 0; r4 < MB / 4; ++r4) {
      float4 v = reinterpret_cast<const float4*>(xp)[r4];
      float vv[4] = {v.x, v.y, v.z, v.w};
      #pragma unroll
      for (int j = 0; j < 4; ++j) {
        int r = r4 * 4 + j;
        bf16 h = (bf16)vv[j];
        bf16 l = (bf16)(vv[j] - (float)h);
        sA_hi[r][tid] = h;
        sA_lo[r][tid] = l;
      }
    }
  }
  __syncthreads();

  // ---- split-bf16 WMMA GEMM: per wave, 32 rows x 128 codes ----
  v8f acc[2][8];
  #pragma unroll
  for (int mt = 0; mt < 2; ++mt)
    #pragma unroll
    for (int ct = 0; ct < 8; ++ct)
      acc[mt][ct] = (v8f){0.f,0.f,0.f,0.f,0.f,0.f,0.f,0.f};

  for (int kb = 0; kb < D_DIM; kb += 32) {
    v16bf ah[2], al[2];
    #pragma unroll
    for (int mt = 0; mt < 2; ++mt) {
      const bf16* ph = &sA_hi[mt * 16 + rc][kb + g * 8];
      ah[mt] = cat8(*reinterpret_cast<const v8bf*>(ph),
                    *reinterpret_cast<const v8bf*>(ph + 16));
      const bf16* pl = &sA_lo[mt * 16 + rc][kb + g * 8];
      al[mt] = cat8(*reinterpret_cast<const v8bf*>(pl),
                    *reinterpret_cast<const v8bf*>(pl + 16));
    }
    #pragma unroll
    for (int ct = 0; ct < 8; ++ct) {
      const int code = wv * 128 + ct * 16 + rc;
      const bf16* pbh = cb_hi + (size_t)code * D_DIM + kb + g * 8;
      const bf16* pbl = cb_lo + (size_t)code * D_DIM + kb + g * 8;
      v16bf bh = cat8(*reinterpret_cast<const v8bf*>(pbh),
                      *reinterpret_cast<const v8bf*>(pbh + 16));
      v16bf bl = cat8(*reinterpret_cast<const v8bf*>(pbl),
                      *reinterpret_cast<const v8bf*>(pbl + 16));
      #pragma unroll
      for (int mt = 0; mt < 2; ++mt) {
        acc[mt][ct] = __builtin_amdgcn_wmma_f32_16x16x32_bf16(
            false, ah[mt], false, bh, (short)0, acc[mt][ct], false, false);
        acc[mt][ct] = __builtin_amdgcn_wmma_f32_16x16x32_bf16(
            false, ah[mt], false, bl, (short)0, acc[mt][ct], false, false);
        acc[mt][ct] = __builtin_amdgcn_wmma_f32_16x16x32_bf16(
            false, al[mt], false, bh, (short)0, acc[mt][ct], false, false);
      }
    }
  }

  // ---- per-lane argmin over distance proxy e^2 - 2*dot ----
  float es[8];
  #pragma unroll
  for (int ct = 0; ct < 8; ++ct) es[ct] = esq[wv * 128 + ct * 16 + rc];

  float bval[2][8];
  int   bidx[2][8];
  #pragma unroll
  for (int mt = 0; mt < 2; ++mt)
    #pragma unroll
    for (int r = 0; r < 8; ++r) { bval[mt][r] = 3.4e38f; bidx[mt][r] = 0; }

  #pragma unroll
  for (int ct = 0; ct < 8; ++ct)
    #pragma unroll
    for (int mt = 0; mt < 2; ++mt) {
      v8f a = acc[mt][ct];
      #pragma unroll
      for (int r = 0; r < 8; ++r) {
        float dd = es[ct] - 2.f * a[r];
        if (dd < bval[mt][r]) { bval[mt][r] = dd; bidx[mt][r] = wv * 128 + ct * 16 + rc; }
      }
    }

  // reduce across the 16 lanes holding the same rows (wave32 halves)
  #pragma unroll
  for (int mask = 1; mask < 16; mask <<= 1) {
    #pragma unroll
    for (int mt = 0; mt < 2; ++mt)
      #pragma unroll
      for (int r = 0; r < 8; ++r) {
        float ov = __shfl_xor(bval[mt][r], mask);
        int   oi = __shfl_xor(bidx[mt][r], mask);
        if (ov < bval[mt][r] || (ov == bval[mt][r] && oi < bidx[mt][r])) {
          bval[mt][r] = ov; bidx[mt][r] = oi;
        }
      }
  }
  if (rc == 0) {
    #pragma unroll
    for (int mt = 0; mt < 2; ++mt)
      #pragma unroll
      for (int r = 0; r < 8; ++r) {
        int m = mt * 16 + r + 8 * g;  // C/D layout: vgpr r, lane-half g -> row
        s_cval[wv][m] = bval[mt][r];
        s_cidx[wv][m] = bidx[mt][r];
      }
  }
  __syncthreads();

  // ---- reduce across the 8 waves, emit indices ----
  if (tid < MB) {
    float bv = s_cval[0][tid];
    int   bi = s_cidx[0][tid];
    #pragma unroll
    for (int w = 1; w < 8; ++w) {
      float ov = s_cval[w][tid]; int oi = s_cidx[w][tid];
      if (ov < bv || (ov == bv && oi < bi)) { bv = ov; bi = oi; }
    }
    s_fidx[tid] = bi;
    out[1 + n0 + tid] = (float)bi;   // encoding_indices
  }
  __syncthreads();

  // ---- gather winning codebook rows, write NCHW output, loss partial ----
  float lsum = 0.f;
  {
    float* qout = out + 1 + N_ROWS + (size_t)b * (D_DIM * HWSZ) + (size_t)tid * HWSZ + hw0;
    #pragma unroll
    for (int r = 0; r < MB; ++r) {
      int id = s_fidx[r];
      float q  = cb[(size_t)id * D_DIM + tid];
      float xv = (float)sA_hi[r][tid] + (float)sA_lo[r][tid];  // ~fp32 x
      float df = q - xv;
      lsum += df * df;
      qout[r] = q;
    }
  }
  #pragma unroll
  for (int m = 16; m >= 1; m >>= 1) lsum += __shfl_xor(lsum, m);
  if (lane == 0) s_red[wv] = lsum;
  __syncthreads();
  if (tid == 0) {
    float tot = 0.f;
    #pragma unroll
    for (int i = 0; i < 8; ++i) tot += s_red[i];
    loss_part[blockIdx.x] = tot;   // deterministic: no atomics
  }
}

// ---------------- deterministic loss finalize ----------------
__global__ __launch_bounds__(256) void vq_loss(const float* __restrict__ part,
                                               float* __restrict__ out) {
  __shared__ float sred[8];
  const int t = threadIdx.x;
  float s = part[t] + part[t + 256] + part[t + 512] + part[t + 768];
  #pragma unroll
  for (int m = 16; m >= 1; m >>= 1) s += __shfl_xor(s, m);
  if ((t & 31) == 0) sred[t >> 5] = s;
  __syncthreads();
  if (t == 0) {
    float tot = 0.f;
    #pragma unroll
    for (int i = 0; i < 8; ++i) tot += sred[i];
    // loss = (1 + beta) * mean((q - x)^2), numel = 32768 * 256
    out[0] = tot * (1.25f / 8388608.f);
  }
}

extern "C" void kernel_launch(void* const* d_in, const int* in_sizes, int n_in,
                              void* d_out, int out_size, void* d_ws, size_t ws_size,
                              hipStream_t stream) {
  const float* x  = (const float*)d_in[0];  // [32,256,32,32]
  const float* cb = (const float*)d_in[1];  // [1024,256]
  char* w = (char*)d_ws;
  bf16*  cb_hi = (bf16*)(w);                                     // 512 KB
  bf16*  cb_lo = (bf16*)(w + (size_t)K_VEC * D_DIM * 2);         // 512 KB
  float* esq   = (float*)(w + (size_t)K_VEC * D_DIM * 4);        // 4 KB
  float* part  = (float*)(w + (size_t)K_VEC * D_DIM * 4 + K_VEC * 4); // 4 KB
  float* out   = (float*)d_out;  // [loss(1) | indices(32768) | quantized(8388608)]

  vq_prep<<<K_VEC, 256, 0, stream>>>(cb, cb_hi, cb_lo, esq);
  vq_main<<<N_ROWS / MB, 256, 0, stream>>>(x, cb, cb_hi, cb_lo, esq, out, part);
  vq_loss<<<1, 256, 0, stream>>>(part, out);
}
